// Bidirectional_LSTM_Decoder_19971597927087
// MI455X (gfx1250) — matrix operand order repaired and checked by reading, not compiled, not run on hardware
//
#include <hip/hip_runtime.h>
#include <hip/hip_bf16.h>

// ---------------------------------------------------------------------------
// Bidirectional 2-layer LSTM decoder for MI455X (gfx1250, wave32, WMMA).
// B=64, T=256, I=H=1024.  bf16 WMMA 16x16x32, f32 accumulate.
// ---------------------------------------------------------------------------

#define B_ 64
#define T_ 256
#define I_ 1024
#define H_ 1024

typedef __attribute__((ext_vector_type(16))) __bf16  v16bf_t;
typedef __attribute__((ext_vector_type(8)))  float   v8f_t;
typedef __attribute__((ext_vector_type(4)))  unsigned int u32x4;

union FragAB { v16bf_t v; u32x4 u[2]; };

__device__ __forceinline__ unsigned short f2bf(float f) {
  unsigned int u = __float_as_uint(f);
  unsigned int r = (u + 0x7FFFu + ((u >> 16) & 1u)) >> 16;  // RNE
  return (unsigned short)r;
}
__device__ __forceinline__ float sigmoidf_(float x) {
  return 1.0f / (1.0f + __expf(-x));
}

struct WPtrs {
  const float* wih[4]; const float* whh[4];
  const float* bih[4]; const float* bhh[4];
};

// ---------------------------------------------------------------------------
// Pack weights into WMMA B-fragment order, bf16.
// Combined K axis: k<1024 -> W_ih[j][k], k>=1024 -> W_hh[j][k-1024].
// Fragment (cell, nT64, gate, nsub, ktG) = 512 bf16 = 1KB:
//   lane l (c16=l&15 column, khalf=l>>4), dword d, pair p ->
//   kk = khalf*16 + d*2 + p  (matches 16-bit B-operand consumption order
//   we use in the GEMM: per-lane contiguous 32B, two b128 loads).
// Flat idx bits: [0]=p [3:1]=d [8:4]=lane [14:9]=kt [16:15]=nsub
//                [18:17]=gate [22:19]=nT64 [24:23]=cell   (total 2^25 elems)
// ---------------------------------------------------------------------------
__global__ __launch_bounds__(256) void pack_weights_kernel(WPtrs p, unsigned short* __restrict__ Wp) {
  long idx = (long)blockIdx.x * 256 + threadIdx.x;
  int pbit = (int)(idx & 1);
  int d    = (int)((idx >> 1) & 7);
  int l    = (int)((idx >> 4) & 31);
  int kt   = (int)((idx >> 9) & 63);
  int nsub = (int)((idx >> 15) & 3);
  int g    = (int)((idx >> 17) & 3);
  int nT   = (int)((idx >> 19) & 15);
  int cell = (int)(idx >> 23);
  int c16 = l & 15, kh = l >> 4;
  int k = kt * 32 + kh * 16 + d * 2 + pbit;
  int j = g * 1024 + nT * 64 + nsub * 16 + c16;
  float v = (k < 1024) ? p.wih[cell][(long)j * 1024 + k]
                       : p.whh[cell][(long)j * 1024 + (k - 1024)];
  Wp[idx] = f2bf(v);
}

__global__ __launch_bounds__(256) void bias_kernel(WPtrs p, float* __restrict__ biasC) {
  int i = blockIdx.x * 256 + threadIdx.x;
  if (i < 4 * 4096) {
    int cell = i >> 12, j = i & 4095;
    biasC[i] = p.bih[cell][j] + p.bhh[cell][j];
  }
}

// Convert input to bf16 (and feature-reversed copy for the "backward" cells).
__global__ __launch_bounds__(256) void pack_x_kernel(const float* __restrict__ in,
                                                     unsigned short* __restrict__ xb,
                                                     unsigned short* __restrict__ xrb) {
  long idx = (long)blockIdx.x * 256 + threadIdx.x;   // B*T*I = 16777216
  long bt = idx >> 10;
  int  k  = (int)(idx & 1023);
  unsigned short v = f2bf(in[idx]);
  xb[idx] = v;
  xrb[bt * 1024 + (1023 - k)] = v;                    // xr = x[:, ::-1]
}

// Initialize h/c state for all 4 cells from encoder_h / encoder_c.
// hbf layout: [cell][parity][B*H]   hf32/cst: [cell][B*H]
__global__ __launch_bounds__(256) void init_state_kernel(const float* __restrict__ eh,
                                                         const float* __restrict__ ec,
                                                         unsigned short* __restrict__ hbf,
                                                         float* __restrict__ hf32,
                                                         float* __restrict__ cst) {
  int i = blockIdx.x * 256 + threadIdx.x;   // 2 * B * H = 131072
  int dir = i >> 16, r = i & 65535;
  int b = r >> 10, col = r & 1023;
  float h = eh[(long)b * 2048 + dir * 1024 + col];
  float c = ec[(long)b * 2048 + dir * 1024 + col];
#pragma unroll
  for (int layer = 0; layer < 2; ++layer) {
    int cell = layer * 2 + dir;
    hf32[cell * 65536 + r] = h;
    cst[cell * 65536 + r]  = c;
    hbf[cell * 131072 + r] = f2bf(h);       // parity 0
  }
}

// ---------------------------------------------------------------------------
// One LSTM step for one layer (both directions).
// Grid: 64 blocks (dir = blockIdx>>5, 32-column tile = blockIdx&31),
// 256 threads = 8 waves. Each wave: mt = wave&3 (16 batch rows),
// one 16-col nsub, all 4 gates -> 4 v8f accumulators, K=2048 in 32-chunks.
// A panel (64 x 128 bf16) staged in LDS per 128-K chunk.
// ---------------------------------------------------------------------------
struct StepArgs {
  const unsigned short* Wp;     // packed weights, all 4 cells
  const float* bias;            // [4][4096] combined biases
  const unsigned short* xF;     // A k<1024 source, dir 0 (row stride xStride)
  const unsigned short* xB;     // A k<1024 source, dir 1
  long xStride;
  unsigned short* hbf;          // [4][2][65536] bf16 h (double-buffered)
  float* hf32;                  // [4][65536]
  float* cst;                   // [4][65536]
  float* out;                   // null for layer 0; (B,T,2H) for layer 1
  int t;
  int cellBase;                 // 0 for layer 0, 2 for layer 1
};

__global__ __launch_bounds__(256) void lstm_step_kernel(StepArgs a) {
  __shared__ __align__(16) unsigned short As[64 * 128];

  const int tid  = threadIdx.x;
  const int wave = tid >> 5;
  const int lane = tid & 31;
  const int dir  = blockIdx.x >> 5;
  const int nT32 = blockIdx.x & 31;
  const int nT64 = nT32 >> 1;
  const int cell = a.cellBase + dir;
  const int mt   = wave & 3;
  const int nsub = (nT32 & 1) * 2 + (wave >> 2);
  const int pr   = a.t & 1, pw = pr ^ 1;

  const unsigned short* __restrict__ xsrc  = dir ? a.xB : a.xF;
  const unsigned short* __restrict__ hprev = a.hbf + cell * 131072 + pr * 65536;
  const unsigned short* __restrict__ wcell = a.Wp + (long)cell * (16L * 4 * 4 * 64 * 512);

  v8f_t acc[4];
  const v8f_t vzero = {};
#pragma unroll
  for (int g = 0; g < 4; ++g) acc[g] = vzero;

  // cooperative A-panel load indices: 256 threads x 64B each (64 rows x 128 K bf16)
  const int prow = tid >> 2;
  const int pcol = (tid & 3) * 32;
  // A-fragment indices (contiguous k order used here, same as B pack: lanes 0-15 K 0-15, lanes 16-31 K 16-31)
  const int arow = mt * 16 + (lane & 15);
  const int kh8  = (lane >> 4) * 16;

  for (int kc = 0; kc < 16; ++kc) {
    if (kc) __syncthreads();
    const int kbase = kc * 128;
    const unsigned short* src = (kbase < 1024)
        ? (xsrc  + (long)prow * a.xStride + kbase + pcol)
        : (hprev + (long)prow * 1024 + (kbase - 1024) + pcol);
    u32x4* dst = (u32x4*)&As[prow * 128 + pcol];
    const u32x4* s4 = (const u32x4*)src;
    dst[0] = s4[0]; dst[1] = s4[1]; dst[2] = s4[2]; dst[3] = s4[3];
    __syncthreads();

#pragma unroll
    for (int kt = 0; kt < 4; ++kt) {
      FragAB afr;
      const u32x4* lp = (const u32x4*)&As[arow * 128 + kt * 32 + kh8];
      afr.u[0] = lp[0];        // K+0..7   (or +16..23 for lanes 16-31)
      afr.u[1] = lp[1];        // K+8..15  (or +24..31)
      const int ktG = kc * 4 + kt;
#pragma unroll
      for (int g = 0; g < 4; ++g) {
        FragAB bfr;
        const long off = ((((long)(nT64 * 4 + g)) * 4 + nsub) * 64 + ktG) * 512 + lane * 16;
        const u32x4* bp = (const u32x4*)(wcell + off);
        bfr.u[0] = bp[0]; bfr.u[1] = bp[1];
        acc[g] = __builtin_amdgcn_wmma_f32_16x16x32_bf16(
            false, afr.v, false, bfr.v, (short)0, acc[g], false, false);
      }
    }
  }

  // Epilogue: gates -> c,h update.  C layout: lanes 0-15 N=lane M=r,
  // lanes 16-31 N=lane-16 M=r+8.
  const int colH = nT64 * 64 + nsub * 16 + (lane & 15);
  const float* biasc = a.bias + cell * 4096;
  const float b_i = biasc[0 * 1024 + colH];
  const float b_f = biasc[1 * 1024 + colH];
  const float b_g = biasc[2 * 1024 + colH];
  const float b_o = biasc[3 * 1024 + colH];
  float* cptr = a.cst  + cell * 65536;
  float* hptr = a.hf32 + cell * 65536;
  unsigned short* hnext = a.hbf + cell * 131072 + pw * 65536;

#pragma unroll
  for (int r = 0; r < 8; ++r) {
    const int b = mt * 16 + r + ((lane >> 4) << 3);
    const long si = (long)b * 1024 + colH;
    float iv = sigmoidf_(acc[0][r] + b_i);
    float fv = sigmoidf_(acc[1][r] + b_f);
    float gv = tanhf(acc[2][r] + b_g);
    float ov = sigmoidf_(acc[3][r] + b_o);
    float cn = fv * cptr[si] + iv * gv;
    float hn = ov * tanhf(cn);
    cptr[si] = cn;
    hptr[si] = hn;
    hnext[si] = f2bf(hn);
    if (a.out) {
      a.out[((long)b * T_ + a.t) * 2048 + dir * 1024 + colH] = hn;
    }
  }
}

// Final h,c outputs: concat(layer-1 fwd, layer-1 bwd).
__global__ __launch_bounds__(256) void finalize_kernel(const float* __restrict__ hf32,
                                                       const float* __restrict__ cst,
                                                       float* __restrict__ out) {
  int i = blockIdx.x * 256 + threadIdx.x;   // 2 * B * 2H = 262144
  int which = i >> 17, r = i & 131071;
  int b = r >> 11, col = r & 2047;
  int cell = 2 + (col >> 10), ch = col & 1023;
  const float* src = which ? cst : hf32;
  out[33554432L + (long)which * 131072 + r] = src[cell * 65536 + (long)b * 1024 + ch];
}

// ---------------------------------------------------------------------------
extern "C" void kernel_launch(void* const* d_in, const int* in_sizes, int n_in,
                              void* d_out, int out_size, void* d_ws, size_t ws_size,
                              hipStream_t stream) {
  (void)in_sizes; (void)n_in; (void)out_size; (void)ws_size;
  const float* input = (const float*)d_in[0];
  const float* enc_h = (const float*)d_in[3];
  const float* enc_c = (const float*)d_in[4];
  WPtrs wp;
  for (int c = 0; c < 4; ++c) {            // order: 0f, 0b, 1f, 1b
    wp.wih[c] = (const float*)d_in[5 + c * 4 + 0];
    wp.whh[c] = (const float*)d_in[5 + c * 4 + 1];
    wp.bih[c] = (const float*)d_in[5 + c * 4 + 2];
    wp.bhh[c] = (const float*)d_in[5 + c * 4 + 3];
  }

  char* ws = (char*)d_ws;
  unsigned short* Wp   = (unsigned short*)(ws);                 //  67,108,864 B
  float*          bias = (float*)(ws + 67108864);               //      65,536 B
  unsigned short* xb   = (unsigned short*)(ws + 67174400);      //  33,554,432 B
  unsigned short* xrb  = (unsigned short*)(ws + 100728832);     //  33,554,432 B
  unsigned short* hbf  = (unsigned short*)(ws + 134283264);     //   1,048,576 B
  float*          hf32 = (float*)(ws + 135331840);              //   1,048,576 B
  float*          cst  = (float*)(ws + 136380416);              //   1,048,576 B
  float* out = (float*)d_out;

  pack_weights_kernel<<<33554432 / 256, 256, 0, stream>>>(wp, Wp);
  bias_kernel<<<64, 256, 0, stream>>>(wp, bias);
  pack_x_kernel<<<16777216 / 256, 256, 0, stream>>>(input, xb, xrb);
  init_state_kernel<<<512, 256, 0, stream>>>(enc_h, enc_c, hbf, hf32, cst);

  for (int t = 0; t < T_; ++t) {
    StepArgs a0;
    a0.Wp = Wp; a0.bias = bias;
    a0.xF = xb  + (long)t * 1024;
    a0.xB = xrb + (long)t * 1024;
    a0.xStride = (long)T_ * 1024;
    a0.hbf = hbf; a0.hf32 = hf32; a0.cst = cst;
    a0.out = nullptr; a0.t = t; a0.cellBase = 0;
    lstm_step_kernel<<<64, 256, 0, stream>>>(a0);

    const int pw = (t + 1) & 1;
    StepArgs a1 = a0;
    a1.xF = hbf + 0 * 131072 + pw * 65536;  // h0f just written this step
    a1.xB = hbf + 1 * 131072 + pw * 65536;  // h0b just written this step
    a1.xStride = 1024;
    a1.out = out; a1.cellBase = 2;
    lstm_step_kernel<<<64, 256, 0, stream>>>(a1);
  }

  finalize_kernel<<<1024, 256, 0, stream>>>(hf32, cst, out);
}
